// GINEncoder2_17205638988407
// MI455X (gfx1250) — matrix operand order repaired and checked by reading, not compiled, not run on hardware
//
#include <hip/hip_runtime.h>
#include <hip/hip_bf16.h>

#define N_NODES 50000
#define N_EDGES 800000
#define FDIM    128
#define NGRAPH  512

typedef __bf16 bf16_t;
typedef __attribute__((ext_vector_type(16))) __bf16 v16bf;
typedef __attribute__((ext_vector_type(8)))  __bf16 v8bf;
typedef __attribute__((ext_vector_type(8)))  float  v8f;

#if defined(__has_builtin)
#  if __has_builtin(__builtin_amdgcn_global_load_async_to_lds_b128) && \
      __has_builtin(__builtin_amdgcn_s_wait_asynccnt)
#    define USE_ASYNC_LDS 1
#  endif
#endif
#ifndef USE_ASYNC_LDS
#  define USE_ASYNC_LDS 0
#endif

typedef int intv4 __attribute__((vector_size(16)));
#if USE_ASYNC_LDS
typedef __attribute__((address_space(1))) intv4 g_v4i;   // global int4
typedef __attribute__((address_space(3))) intv4 l_v4i;   // LDS int4
#endif

__device__ __forceinline__ bf16_t f2bf(float f) {
    unsigned u = __float_as_uint(f);
    unsigned r = u + 0x7FFFu + ((u >> 16) & 1u);   // round-to-nearest-even
    unsigned short s = (unsigned short)(r >> 16);
    return __builtin_bit_cast(bf16_t, s);
}

__device__ __forceinline__ float sigmoidf_(float x) { return 1.0f / (1.0f + __expf(-x)); }

// ---------------- utility kernels ----------------
__global__ void zero_f32(float* p, int n) {
    int i = blockIdx.x * blockDim.x + threadIdx.x;
    if (i < n) p[i] = 0.0f;
}
__global__ void zero_u16(unsigned short* p, int n) {
    int i = blockIdx.x * blockDim.x + threadIdx.x;
    if (i < n) p[i] = 0;
}
__global__ void cvt_bf16_k(const float* __restrict__ in, bf16_t* __restrict__ out, int n) {
    int i = blockIdx.x * blockDim.x + threadIdx.x;
    if (i < n) out[i] = f2bf(in[i]);
}

// ---------------- WMMA bf16 GEMM ----------------
// D[M,Nout] = A[M,KT] * W[Nout,KT]^T + bias (+D if ACC), optional RELU,
// compile-time selected f32 (OUTF) and/or bf16 (OUTB) outputs.
// W is staged into LDS once per workgroup (async global->LDS when available);
// each wave keeps all KT/32 A-fragments resident and sweeps every n-tile,
// preloading all B-fragments of a tile before the WMMA chain.
template <int KT, bool ACC, bool RELU, bool OUTF, bool OUTB>
__global__ void wmma_gemm_bf16(const bf16_t* __restrict__ A,
                               const bf16_t* __restrict__ Wg,
                               const float*  __restrict__ bias,
                               float* __restrict__ D,
                               bf16_t* __restrict__ Dbf,
                               int M, int Nout) {
    extern __shared__ char smem_raw[];
    bf16_t* sW = (bf16_t*)smem_raw;

    const int tid = threadIdx.x;
    const int tilesM = M >> 4;
    const int tilesN = Nout >> 4;

    // ---- stage W[Nout,KT] (row-major) into LDS, 16B chunks ----
    const int nChunks = (Nout * KT) / 8;          // 8 bf16 = 16 bytes per chunk
#if USE_ASYNC_LDS
    for (int i = tid; i < nChunks; i += blockDim.x) {
        __builtin_amdgcn_global_load_async_to_lds_b128(
            (g_v4i*)(Wg + (size_t)i * 8),
            (l_v4i*)(sW + (size_t)i * 8),
            0, 0);
    }
    __builtin_amdgcn_s_wait_asynccnt(0);
#else
    for (int i = tid; i < nChunks; i += blockDim.x)
        *(uint4*)(sW + (size_t)i * 8) = *(const uint4*)(Wg + (size_t)i * 8);
#endif
    __syncthreads();

    const int tileM = blockIdx.x * (blockDim.x >> 5) + (tid >> 5);
    if (tileM >= tilesM) return;                  // after barrier; uniform per wave

    const int lane = tid & 31;
    const int half = lane >> 4;                   // 0: lanes 0-15, 1: lanes 16-31
    const int l16  = lane & 15;

    // A fragments for the whole K dimension, resident in VGPRs.
    // 16-bit A layout: lanes 0-15 hold K {0..7,16..23}; lanes 16-31 {8..15,24..31}
    const bf16_t* Arow = A + (size_t)(tileM * 16 + l16) * KT;
    v16bf afrag[KT / 32];
    #pragma unroll
    for (int j = 0; j < KT / 32; ++j) {
        v8bf alo = *(const v8bf*)(Arow + j * 32 + 8 * half);
        v8bf ahi = *(const v8bf*)(Arow + j * 32 + 16 + 8 * half);
        afrag[j] = __builtin_shufflevector(alo, ahi,
                                           0,1,2,3,4,5,6,7,8,9,10,11,12,13,14,15);
    }

    const int row0 = tileM * 16 + 8 * half;       // C/D: VGPR r -> row row0+r

    for (int tn = 0; tn < tilesN; ++tn) {
        const int col = tn * 16 + l16;
        const bf16_t* Wrow = sW + (size_t)col * KT + 8 * half;

        // Preload all B fragments for this tile (batched DS loads).
        // B 32x16: lanes 0-15 K {0..7,16..23}, lanes 16-31 K {8..15,24..31} (from LDS)
        v16bf bfrag[KT / 32];
        #pragma unroll
        for (int j = 0; j < KT / 32; ++j) {
            v8bf blo = *(const v8bf*)(Wrow + j * 32);
            v8bf bhi = *(const v8bf*)(Wrow + j * 32 + 16);
            bfrag[j] = __builtin_shufflevector(blo, bhi,
                                               0,1,2,3,4,5,6,7,8,9,10,11,12,13,14,15);
        }

        v8f c;
        if (ACC) {
            #pragma unroll
            for (int r = 0; r < 8; ++r) c[r] = D[(size_t)(row0 + r) * Nout + col];
        } else {
            #pragma unroll
            for (int r = 0; r < 8; ++r) c[r] = 0.0f;
        }

        #pragma unroll
        for (int j = 0; j < KT / 32; ++j)
            c = __builtin_amdgcn_wmma_f32_16x16x32_bf16(false, afrag[j], false, bfrag[j],
                                                        (short)0, c, false, false);

        float bb = bias[col];
        #pragma unroll
        for (int r = 0; r < 8; ++r) {
            float v = c[r] + bb;
            if (RELU) v = fmaxf(v, 0.0f);
            size_t o = (size_t)(row0 + r) * Nout + col;
            if (OUTF) D[o]   = v;
            if (OUTB) Dbf[o] = f2bf(v);
        }
    }
}

// ---------------- graph scatter-add: z[dst] += h[src], one wave per edge ----------------
__global__ void scatter_add_k(const float* __restrict__ h,
                              const int* __restrict__ src,
                              const int* __restrict__ dst,
                              float* __restrict__ z, int E) {
    int e = blockIdx.x * (blockDim.x >> 5) + (threadIdx.x >> 5);
    int lane = threadIdx.x & 31;
    if (e >= E) return;
    int s = src[e], d = dst[e];
    const float* sp = h + (size_t)s * FDIM;
    float* dp = z + (size_t)d * FDIM;
    #pragma unroll
    for (int i = 0; i < FDIM / 32; ++i)
        atomicAdd(dp + lane + 32 * i, sp[lane + 32 * i]);
}

// ---------------- GRU elementwise: updates h (f32) and hbf (bf16) ----------------
__global__ void gru_cell_k(const float* __restrict__ gx, const float* __restrict__ gh,
                           float* __restrict__ h, bf16_t* __restrict__ hbf, int N) {
    int idx = blockIdx.x * blockDim.x + threadIdx.x;
    if (idx >= N * FDIM) return;
    int n = idx / FDIM, d = idx - n * FDIM;
    size_t b3 = (size_t)n * 3 * FDIM;
    float gxr = gx[b3 + d], gxz = gx[b3 + FDIM + d], gxn = gx[b3 + 2 * FDIM + d];
    float ghr = gh[b3 + d], ghz = gh[b3 + FDIM + d], ghn = gh[b3 + 2 * FDIM + d];
    float r  = sigmoidf_(gxr + ghr);
    float zt = sigmoidf_(gxz + ghz);
    float nn = tanhf(gxn + r * ghn);
    float hv = h[idx];
    float ho = (1.0f - zt) * nn + zt * hv;
    h[idx] = ho;
    hbf[idx] = f2bf(ho);
}

// ---------------- LSTM elementwise (gate order i,f,g,o) ----------------
__global__ void lstm_cell_k(const float* __restrict__ gates,
                            float* __restrict__ cl, float* __restrict__ hl,
                            bf16_t* __restrict__ hlbf, int G) {
    int idx = blockIdx.x * blockDim.x + threadIdx.x;
    if (idx >= G * FDIM) return;
    int g = idx / FDIM, d = idx - g * FDIM;
    size_t b4 = (size_t)g * 4 * FDIM;
    float gi = gates[b4 + d], gf = gates[b4 + FDIM + d];
    float gg = gates[b4 + 2 * FDIM + d], go = gates[b4 + 3 * FDIM + d];
    float cv = sigmoidf_(gf) * cl[idx] + sigmoidf_(gi) * tanhf(gg);
    cl[idx] = cv;
    float hv = sigmoidf_(go) * tanhf(cv);
    hl[idx] = hv;
    hlbf[idx] = f2bf(hv);
}

// ---------------- attention ----------------
__global__ void attn_e_k(const float* __restrict__ h, const float* __restrict__ q,
                         const int* __restrict__ batch, float* __restrict__ e, int N) {
    int n = blockIdx.x * blockDim.x + threadIdx.x;
    if (n >= N) return;
    int b = batch[n];
    const float4* hp = (const float4*)(h + (size_t)n * FDIM);
    const float4* qp = (const float4*)(q + (size_t)b * FDIM);
    float s = 0.0f;
    #pragma unroll
    for (int i = 0; i < FDIM / 4; ++i) {
        float4 a = hp[i], c = qp[i];
        s += a.x * c.x + a.y * c.y + a.z * c.z + a.w * c.w;
    }
    e[n] = s;
}

__device__ __forceinline__ unsigned ford(float f) {
    unsigned u = __float_as_uint(f);
    return (u & 0x80000000u) ? ~u : (u | 0x80000000u);
}
__device__ __forceinline__ float unford(unsigned u) {
    unsigned v = (u & 0x80000000u) ? (u & 0x7FFFFFFFu) : ~u;
    return __uint_as_float(v);
}

__global__ void seg_max_k(const float* __restrict__ e, const int* __restrict__ batch,
                          unsigned* __restrict__ emaxu, int N) {
    int n = blockIdx.x * blockDim.x + threadIdx.x;
    if (n >= N) return;
    atomicMax(emaxu + batch[n], ford(e[n]));
}

__global__ void finalize_emax_k(const unsigned* __restrict__ emaxu,
                                float* __restrict__ emaxf, int G) {
    int g = blockIdx.x * blockDim.x + threadIdx.x;
    if (g >= G) return;
    unsigned u = emaxu[g];
    float v = (u == 0u) ? 0.0f : unford(u);
    if (!isfinite(v)) v = 0.0f;
    emaxf[g] = v;
}

__global__ void attn_exp_k(const float* __restrict__ e, const float* __restrict__ emax,
                           const int* __restrict__ batch, float* __restrict__ anum,
                           float* __restrict__ denom, int N) {
    int n = blockIdx.x * blockDim.x + threadIdx.x;
    if (n >= N) return;
    int b = batch[n];
    float v = __expf(e[n] - emax[b]);
    anum[n] = v;
    atomicAdd(denom + b, v);
}

__global__ void attn_rvec_k(const float* __restrict__ h, const float* __restrict__ anum,
                            const float* __restrict__ denom, const int* __restrict__ batch,
                            float* __restrict__ rvec, int N) {
    int n = blockIdx.x * (blockDim.x >> 5) + (threadIdx.x >> 5);
    int lane = threadIdx.x & 31;
    if (n >= N) return;
    int b = batch[n];
    float coef = anum[n] / (denom[b] + 1e-16f);
    const float* hp = h + (size_t)n * FDIM;
    float* rp = rvec + (size_t)b * FDIM;
    #pragma unroll
    for (int i = 0; i < FDIM / 32; ++i)
        atomicAdd(rp + lane + 32 * i, coef * hp[lane + 32 * i]);
}

__global__ void build_qstar_k(const float* __restrict__ hl, const float* __restrict__ rvec,
                              float* __restrict__ qstar, bf16_t* __restrict__ qbf, int G) {
    int idx = blockIdx.x * blockDim.x + threadIdx.x;
    if (idx >= G * 2 * FDIM) return;
    int g = idx / (2 * FDIM), d = idx - g * 2 * FDIM;
    float v = (d < FDIM) ? hl[(size_t)g * FDIM + d]
                         : rvec[(size_t)g * FDIM + (d - FDIM)];
    qstar[idx] = v;
    qbf[idx] = f2bf(v);
}

// ---------------- host driver ----------------
extern "C" void kernel_launch(void* const* d_in, const int* in_sizes, int n_in,
                              void* d_out, int out_size, void* d_ws, size_t ws_size,
                              hipStream_t stream) {
    const int N = N_NODES, E = N_EDGES, G = NGRAPH;

    const float* x        = (const float*)d_in[0];
    const int*   ei       = (const int*)d_in[1];
    const int*   batch    = (const int*)d_in[2];
    const float* lin0_W   = (const float*)d_in[3];
    const float* lin0_b   = (const float*)d_in[4];
    const float* gin_W1   = (const float*)d_in[5];
    const float* gin_b1   = (const float*)d_in[6];
    const float* gin_W2   = (const float*)d_in[7];
    const float* gin_b2   = (const float*)d_in[8];
    const float* gru_Wih  = (const float*)d_in[9];
    const float* gru_Whh  = (const float*)d_in[10];
    const float* gru_bih  = (const float*)d_in[11];
    const float* gru_bhh  = (const float*)d_in[12];
    const float* lstm_Wih = (const float*)d_in[13];
    const float* lstm_Whh = (const float*)d_in[14];
    const float* lstm_bih = (const float*)d_in[15];
    const float* lstm_bhh = (const float*)d_in[16];
    const int* src = ei;
    const int* dst = ei + E;

    // workspace carve
    char* ws = (char*)d_ws;
    size_t off = 0;
    auto carve = [&](size_t bytes) -> void* {
        void* p = ws + off;
        off = (off + bytes + 255) & ~(size_t)255;
        return p;
    };
    float*  bufA  = (float*)carve((size_t)N * FDIM * 4);        // h (f32)
    float*  bufB  = (float*)carve((size_t)N * FDIM * 4);        // z (f32)
    float*  gx    = (float*)carve((size_t)N * 3 * FDIM * 4);
    float*  gh    = (float*)carve((size_t)N * 3 * FDIM * 4);
    bf16_t* bfact = (bf16_t*)carve((size_t)N * FDIM * 2);       // x / z / m (bf16)
    bf16_t* tbf   = (bf16_t*)carve((size_t)N * FDIM * 2);       // t (bf16)
    bf16_t* hbf   = (bf16_t*)carve((size_t)N * FDIM * 2);       // h (bf16)
    bf16_t* lin0_Wb   = (bf16_t*)carve((size_t)FDIM * FDIM * 2);
    bf16_t* gin_W1b   = (bf16_t*)carve((size_t)FDIM * FDIM * 2);
    bf16_t* gin_W2b   = (bf16_t*)carve((size_t)FDIM * FDIM * 2);
    bf16_t* gru_Wihb  = (bf16_t*)carve((size_t)3 * FDIM * FDIM * 2);
    bf16_t* gru_Whhb  = (bf16_t*)carve((size_t)3 * FDIM * FDIM * 2);
    bf16_t* lstm_Wihb = (bf16_t*)carve((size_t)4 * FDIM * 2 * FDIM * 2);
    bf16_t* lstm_Whhb = (bf16_t*)carve((size_t)4 * FDIM * FDIM * 2);
    float*  qstar = (float*)carve((size_t)G * 2 * FDIM * 4);
    bf16_t* qbf   = (bf16_t*)carve((size_t)G * 2 * FDIM * 2);
    float*  hl    = (float*)carve((size_t)G * FDIM * 4);
    bf16_t* hlbf  = (bf16_t*)carve((size_t)G * FDIM * 2);
    float*  cl    = (float*)carve((size_t)G * FDIM * 4);
    float*  gates = (float*)carve((size_t)G * 4 * FDIM * 4);
    float*  ebuf  = (float*)carve((size_t)N * 4);
    float*  anum  = (float*)carve((size_t)N * 4);
    unsigned* emaxu = (unsigned*)carve((size_t)G * 4);
    float*  emaxf = (float*)carve((size_t)G * 4);
    float*  denom = (float*)carve((size_t)G * 4);
    float*  rvec  = (float*)carve((size_t)G * FDIM * 4);

    auto cvt = [&](const float* in, bf16_t* out, int n) {
        cvt_bf16_k<<<(n + 255) / 256, 256, 0, stream>>>(in, out, n);
    };
    auto zero = [&](float* p, int n) {
        zero_f32<<<(n + 255) / 256, 256, 0, stream>>>(p, n);
    };
    auto zerobf = [&](bf16_t* p, int n) {
        zero_u16<<<(n + 255) / 256, 256, 0, stream>>>((unsigned short*)p, n);
    };

    // weight conversions (tiny)
    cvt(lin0_W, lin0_Wb, FDIM * FDIM);
    cvt(gin_W1, gin_W1b, FDIM * FDIM);
    cvt(gin_W2, gin_W2b, FDIM * FDIM);
    cvt(gru_Wih, gru_Wihb, 3 * FDIM * FDIM);
    cvt(gru_Whh, gru_Whhb, 3 * FDIM * FDIM);
    cvt(lstm_Wih, lstm_Wihb, 4 * FDIM * 2 * FDIM);
    cvt(lstm_Whh, lstm_Whhb, 4 * FDIM * FDIM);

    const int blocksN = ((N >> 4) + 7) / 8;   // 8 waves/block, 16 rows/wave
    const int blocksG = ((G >> 4) + 7) / 8;

    // lin0: h = relu(x @ W0^T + b0)   -> f32 (bufA) + bf16 (hbf)
    cvt(x, bfact, N * FDIM);
    wmma_gemm_bf16<128, false, true, true, true>
        <<<blocksN, 256, (size_t)FDIM * 128 * 2, stream>>>(
            bfact, lin0_Wb, lin0_b, bufA, hbf, N, FDIM);

    // 3 GIN + GRU layers
    for (int l = 0; l < 3; ++l) {
        (void)hipMemcpyAsync(bufB, bufA, (size_t)N * FDIM * 4,
                             hipMemcpyDeviceToDevice, stream);
        scatter_add_k<<<(E + 7) / 8, 256, 0, stream>>>(bufA, src, dst, bufB, E); // z
        cvt(bufB, bfact, N * FDIM);                                              // zbf
        // t = relu(z @ W1^T + b1) -> bf16 only
        wmma_gemm_bf16<128, false, true, false, true>
            <<<blocksN, 256, (size_t)FDIM * 128 * 2, stream>>>(
                bfact, gin_W1b, gin_b1, nullptr, tbf, N, FDIM);
        // m = relu(t @ W2^T + b2) -> bf16 only
        wmma_gemm_bf16<128, false, true, false, true>
            <<<blocksN, 256, (size_t)FDIM * 128 * 2, stream>>>(
                tbf, gin_W2b, gin_b2, nullptr, bfact, N, FDIM);
        // gx = m @ Wih^T + bih
        wmma_gemm_bf16<128, false, false, true, false>
            <<<blocksN, 256, (size_t)3 * FDIM * 128 * 2, stream>>>(
                bfact, gru_Wihb, gru_bih, gx, nullptr, N, 3 * FDIM);
        // gh = h @ Whh^T + bhh
        wmma_gemm_bf16<128, false, false, true, false>
            <<<blocksN, 256, (size_t)3 * FDIM * 128 * 2, stream>>>(
                hbf, gru_Whhb, gru_bhh, gh, nullptr, N, 3 * FDIM);
        gru_cell_k<<<(N * FDIM + 255) / 256, 256, 0, stream>>>(gx, gh, bufA, hbf, N);
    }

    // Set2Set
    zero(qstar, G * 2 * FDIM);
    zerobf(qbf, G * 2 * FDIM);
    zero(hl, G * FDIM);
    zerobf(hlbf, G * FDIM);
    zero(cl, G * FDIM);
    for (int s = 0; s < 3; ++s) {
        // gates = q_star @ Wih^T + bih
        wmma_gemm_bf16<256, false, false, true, false>
            <<<blocksG, 256, (size_t)4 * FDIM * 256 * 2, stream>>>(
                qbf, lstm_Wihb, lstm_bih, gates, nullptr, G, 4 * FDIM);
        // gates += hl @ Whh^T + bhh
        wmma_gemm_bf16<128, true, false, true, false>
            <<<blocksG, 256, (size_t)4 * FDIM * 128 * 2, stream>>>(
                hlbf, lstm_Whhb, lstm_bhh, gates, nullptr, G, 4 * FDIM);
        lstm_cell_k<<<(G * FDIM + 255) / 256, 256, 0, stream>>>(gates, cl, hl, hlbf, G);

        zero((float*)emaxu, G);
        zero(denom, G);
        zero(rvec, G * FDIM);
        attn_e_k<<<(N + 255) / 256, 256, 0, stream>>>(bufA, hl, batch, ebuf, N);
        seg_max_k<<<(N + 255) / 256, 256, 0, stream>>>(ebuf, batch, emaxu, N);
        finalize_emax_k<<<(G + 255) / 256, 256, 0, stream>>>(emaxu, emaxf, G);
        attn_exp_k<<<(N + 255) / 256, 256, 0, stream>>>(ebuf, emaxf, batch, anum, denom, N);
        attn_rvec_k<<<(N + 7) / 8, 256, 0, stream>>>(bufA, anum, denom, batch, rvec, N);
        build_qstar_k<<<(G * 2 * FDIM + 255) / 256, 256, 0, stream>>>(hl, rvec, qstar, qbf, G);
    }

    // outputs: (q_star, out) flattened
    (void)hipMemcpyAsync(d_out, qstar, (size_t)G * 2 * FDIM * 4,
                         hipMemcpyDeviceToDevice, stream);
    (void)hipMemcpyAsync((float*)d_out + G * 2 * FDIM, bufA, (size_t)N * FDIM * 4,
                         hipMemcpyDeviceToDevice, stream);
}
